// MainGNN_14362370638529
// MI455X (gfx1250) — matrix operand order repaired and checked by reading, not compiled, not run on hardware
//
#include <hip/hip_runtime.h>

#define N_NODES 20000
#define N_EDGES 320000

typedef __bf16 bf16_t;
typedef bf16_t v16bf __attribute__((ext_vector_type(16)));
typedef float  v8f   __attribute__((ext_vector_type(8)));
typedef unsigned short u16;
typedef u16 v8u  __attribute__((ext_vector_type(8)));
typedef u16 v16u __attribute__((ext_vector_type(16)));
typedef unsigned int u32;
typedef u32 su4 __attribute__((ext_vector_type(4)));
typedef u32 su8 __attribute__((ext_vector_type(8)));

__device__ __forceinline__ u16 f2bf(float f) {
  union { float f; u32 u; } v; v.f = f;
  u32 r = v.u + 0x7FFFu + ((v.u >> 16) & 1u);   // round-to-nearest-even
  return (u16)(r >> 16);
}

__device__ __forceinline__ void atomAddF(float* p, float v) {
  __hip_atomic_fetch_add(p, v, __ATOMIC_RELAXED, __HIP_MEMORY_SCOPE_AGENT);
}

// 1-D contiguous Tensor-Data-Mover copy: global -> LDS (nbytes multiple of 8).
// D# group0/group1 per CDNA5 ISA ch.8; tracked by TENSORcnt.
__device__ __forceinline__ void tdm_load_1d(void* lds_dst, const void* gsrc, u32 nbytes) {
  const u32 units = nbytes >> 3;                       // 8-byte elements
  unsigned long long ga = (unsigned long long)(uintptr_t)gsrc;
  su4 g0;
  g0[0] = 1u;                                          // count=1, user mode
  g0[1] = (u32)(uintptr_t)lds_dst;                     // lds_addr (LDS byte offset)
  g0[2] = (u32)ga;                                     // global_addr[31:0]
  g0[3] = ((u32)(ga >> 32) & 0x01FFFFFFu) | (2u << 30);// global_addr[56:32] | type=2
  su8 g1;
  g1[0] = (3u << 16);                                  // wg_mask=0, data_size=3 (8B)
  g1[1] = (units & 0xFFFFu) << 16;                     // tensor_dim0[15:0]
  g1[2] = (units >> 16) & 0xFFFFu;                     // tensor_dim0[31:16], tensor_dim1=0
  g1[3] = (units & 0xFFFFu) << 16;                     // tile_dim0
  g1[4] = 0u;                                          // tile_dim1=0, tile_dim2=0 (1-D)
  g1[5] = units;                                       // tensor_dim0_stride (unused for 1-D)
  g1[6] = 0u;
  g1[7] = 0u;
  asm volatile("tensor_load_to_lds %0, %1" :: "s"(g0), "s"(g1) : "memory");
}

__global__ void zero_kernel(float* __restrict__ p, int n) {
  int i = blockIdx.x * blockDim.x + threadIdx.x;
  if (i < n) p[i] = 0.0f;
}

__global__ void count_kernel(const int* __restrict__ dst, float* __restrict__ cnt) {
  int e = blockIdx.x * blockDim.x + threadIdx.x;
  if (e < N_EDGES) atomAddF(&cnt[dst[e]], 1.0f);
}

// w2: [64, C] f32 row-major  ->  w2t: [C, 64] bf16 row-major
__global__ void wprep_kernel(const float* __restrict__ w2, u16* __restrict__ w2t, int C) {
  int idx = blockIdx.x * blockDim.x + threadIdx.x;
  if (idx >= C * 64) return;
  int c = idx >> 6, m = idx & 63;
  w2t[idx] = f2bf(w2[m * C + c]);
}

// hid[e,m] = relu(ea[e,:2] @ w1[2,64] + b1)  stored bf16 [E,64]
__global__ void edge_mlp_kernel(const float* __restrict__ ea,
                                const float* __restrict__ w1,
                                const float* __restrict__ b1,
                                u16* __restrict__ hid) {
  int idx = blockIdx.x * blockDim.x + threadIdx.x;
  if (idx >= N_EDGES * 64) return;
  int e = idx >> 6, m = idx & 63;
  float v = fmaf(ea[e * 2], w1[m], fmaf(ea[e * 2 + 1], w1[64 + m], b1[m]));
  hid[idx] = f2bf(fmaxf(v, 0.0f));
}

// Fused: W_e = hid @ w2 + b2  (WMMA bf16, tile [16 edges,16 cols]);
// msg[e,o] = sum_i feat[src[e],i] * W_e[i,o]; scatter-add to summed[dst].
template<int IN_DIM>
__global__ __launch_bounds__(128)
void msg_kernel(const float* __restrict__ feat,   // [N_NODES, IN_DIM] f32
                const u16*  __restrict__ hid,     // [N_EDGES, 64] bf16
                const u16*  __restrict__ w2t,     // [IN_DIM*32, 64] bf16 (pre-transposed)
                const float* __restrict__ b2,     // [IN_DIM*32]
                const int*  __restrict__ src,
                const int*  __restrict__ dst,
                float* __restrict__ summed)       // [N_NODES, 32]
{
  constexpr int COLS = IN_DIM * 32;
  __shared__ u16   sw[COLS * 64];      // staged w2t
  __shared__ float sb2[COLS];
  __shared__ float sh[4][16 * IN_DIM]; // per-wave gathered source features

  const int tid = threadIdx.x;
  // TDM: DMA w2t (COLS*128 bytes, contiguous) into LDS; wave 0 issues + waits.
  if (tid < 32) {
    tdm_load_1d(sw, w2t, (u32)COLS * 128u);
    __builtin_amdgcn_s_wait_tensorcnt(0);
  }
  for (int t = tid; t < COLS; t += 128) sb2[t] = b2[t];
  __syncthreads();

  const int lane    = tid & 31;
  const int wave    = tid >> 5;
  const int half    = lane >> 4;
  const int ln      = lane & 15;
  const int rowbase = half * 8;    // C/D row base for this lane half
  const int kbaseA  = half * 8;    // A K base (interleaved layout)
  const int kbaseB  = half * 8;    // B K base (interleaved layout)
  float* mysh = &sh[wave][0];

  const int ntiles = N_EDGES / 16;
  const int nwaves = gridDim.x * 4;
  for (int tile = blockIdx.x * 4 + wave; tile < ntiles; tile += nwaves) {
    const int ebase = tile * 16;

    // gather 16 source-feature rows into LDS
    for (int t = lane; t < 16 * IN_DIM; t += 32) {
      const int el = t / IN_DIM;
      const int f  = t - el * IN_DIM;
      mysh[t] = feat[(size_t)src[ebase + el] * IN_DIM + f];
    }

    // A matrix: hid rows (K=64, two 16x16x32 k-chunks). bf16 A layout:
    // lanes 0-15: K {0..7,16..23}; lanes 16-31: K {8..15,24..31} (+32 for chunk 1)
    const u16* hp = hid + (size_t)(ebase + ln) * 64 + kbaseA;
    v16u a0 = __builtin_shufflevector(*(const v8u*)(hp),      *(const v8u*)(hp + 16),
                                      0,1,2,3,4,5,6,7,8,9,10,11,12,13,14,15);
    v16u a1 = __builtin_shufflevector(*(const v8u*)(hp + 32), *(const v8u*)(hp + 48),
                                      0,1,2,3,4,5,6,7,8,9,10,11,12,13,14,15);
    const v16bf A0 = __builtin_bit_cast(v16bf, a0);
    const v16bf A1 = __builtin_bit_cast(v16bf, a1);

    // prefetch next tile's hid rows
    if (tile + nwaves < ntiles)
      __builtin_prefetch(hid + (size_t)((tile + nwaves) * 16 + ln) * 64, 0, 1);

    float m0[8], m1[8];
    #pragma unroll
    for (int r = 0; r < 8; ++r) { m0[r] = 0.0f; m1[r] = 0.0f; }

    for (int i = 0; i < IN_DIM; ++i) {
      float s[8];
      #pragma unroll
      for (int r = 0; r < 8; ++r)
        s[r] = mysh[(rowbase + r) * IN_DIM + i];

      #pragma unroll
      for (int oh = 0; oh < 2; ++oh) {
        const int col = i * 32 + oh * 16 + ln;
        const u16* wp = &sw[col * 64];
        // B layout: lane N=ln, lanes 0-15: K {0..7,16..23}; lanes 16-31: K {8..15,24..31} (+32 for chunk 1)
        v16u b0 = __builtin_shufflevector(*(const v8u*)(wp + kbaseB),
                                          *(const v8u*)(wp + kbaseB + 16),
                                          0,1,2,3,4,5,6,7,8,9,10,11,12,13,14,15);
        v16u b1 = __builtin_shufflevector(*(const v8u*)(wp + 32 + kbaseB),
                                          *(const v8u*)(wp + 48 + kbaseB),
                                          0,1,2,3,4,5,6,7,8,9,10,11,12,13,14,15);
        v8f acc = {};
        acc = __builtin_amdgcn_wmma_f32_16x16x32_bf16(
                false, A0, false, __builtin_bit_cast(v16bf, b0), (short)0, acc, false, false);
        acc = __builtin_amdgcn_wmma_f32_16x16x32_bf16(
                false, A1, false, __builtin_bit_cast(v16bf, b1), (short)0, acc, false, false);
        const float bv = sb2[col];
        float* m = oh ? m1 : m0;
        #pragma unroll
        for (int r = 0; r < 8; ++r)
          m[r] = fmaf(s[r], acc[r] + bv, m[r]);
      }
    }

    // scatter messages to destination nodes
    #pragma unroll
    for (int r = 0; r < 8; ++r) {
      const int d = dst[ebase + rowbase + r];
      float* p = summed + (size_t)d * 32 + ln;
      atomAddF(p,      m0[r]);
      atomAddF(p + 16, m1[r]);
    }
  }
}

// h = relu(summed/max(cnt,1) + x @ c1_root + bias)
__global__ void node1_kernel(const float* __restrict__ summed, const float* __restrict__ cnt,
                             const float* __restrict__ x, const float* __restrict__ root,
                             const float* __restrict__ bias, float* __restrict__ h) {
  int idx = blockIdx.x * blockDim.x + threadIdx.x;
  if (idx >= N_NODES * 32) return;
  int n = idx >> 5, c = idx & 31;
  float v = summed[idx] / fmaxf(cnt[n], 1.0f) + bias[c];
  #pragma unroll
  for (int j = 0; j < 3; ++j) v = fmaf(x[n * 3 + j], root[j * 32 + c], v);
  h[idx] = fmaxf(v, 0.0f);
}

// h2 = relu(summed/max(cnt,1) + h @ c2_root + bias); out = h2 @ fc_w + fc_b
__global__ void node2_kernel(const float* __restrict__ summed, const float* __restrict__ cnt,
                             const float* __restrict__ h, const float* __restrict__ root,
                             const float* __restrict__ bias,
                             const float* __restrict__ fcw, const float* __restrict__ fcb,
                             float* __restrict__ out) {
  int n = blockIdx.x * blockDim.x + threadIdx.x;
  if (n >= N_NODES) return;
  float hr[32];
  #pragma unroll
  for (int j = 0; j < 32; ++j) hr[j] = h[n * 32 + j];
  const float inv = 1.0f / fmaxf(cnt[n], 1.0f);
  float o0 = fcb[0], o1 = fcb[1];
  for (int c = 0; c < 32; ++c) {
    float v = fmaf(summed[n * 32 + c], inv, bias[c]);
    #pragma unroll
    for (int j = 0; j < 32; ++j) v = fmaf(hr[j], root[j * 32 + c], v);
    v = fmaxf(v, 0.0f);
    o0 = fmaf(v, fcw[c * 2 + 0], o0);
    o1 = fmaf(v, fcw[c * 2 + 1], o1);
  }
  out[n * 2 + 0] = o0;
  out[n * 2 + 1] = o1;
}

extern "C" void kernel_launch(void* const* d_in, const int* in_sizes, int n_in,
                              void* d_out, int out_size, void* d_ws, size_t ws_size,
                              hipStream_t stream) {
  (void)in_sizes; (void)n_in; (void)out_size; (void)ws_size;
  const float* x    = (const float*)d_in[0];
  const int*   ei   = (const int*)d_in[1];
  const float* ea   = (const float*)d_in[2];
  const float* e1w1 = (const float*)d_in[3];
  const float* e1b1 = (const float*)d_in[4];
  const float* e1w2 = (const float*)d_in[5];
  const float* e1b2 = (const float*)d_in[6];
  const float* c1r  = (const float*)d_in[7];
  const float* c1b  = (const float*)d_in[8];
  const float* e2w1 = (const float*)d_in[9];
  const float* e2b1 = (const float*)d_in[10];
  const float* e2w2 = (const float*)d_in[11];
  const float* e2b2 = (const float*)d_in[12];
  const float* c2r  = (const float*)d_in[13];
  const float* c2b  = (const float*)d_in[14];
  const float* fcw  = (const float*)d_in[15];
  const float* fcb  = (const float*)d_in[16];
  const int* src = ei;
  const int* dst = ei + N_EDGES;
  float* out = (float*)d_out;

  char* ws = (char*)d_ws;
  size_t off = 0;
  auto alloc = [&](size_t bytes) -> void* {
    void* p = ws + off;
    off += (bytes + 255) & ~(size_t)255;
    return p;
  };
  float* summed1 = (float*)alloc((size_t)N_NODES * 32 * 4);  // 2.56 MB (mult of 256)
  float* summed2 = (float*)alloc((size_t)N_NODES * 32 * 4);  // contiguous with summed1
  float* cnt     = (float*)alloc((size_t)N_NODES * 4);       // contiguous with summed2
  float* h       = (float*)alloc((size_t)N_NODES * 32 * 4);
  u16*   hid     = (u16*)  alloc((size_t)N_EDGES * 64 * 2);  // 41 MB, reused per layer
  u16*   w2t1    = (u16*)  alloc((size_t)96 * 64 * 2);
  u16*   w2t2    = (u16*)  alloc((size_t)1024 * 64 * 2);

  // zero accumulators (summed1|summed2|cnt are contiguous)
  const int nz = N_NODES * 65;
  zero_kernel<<<(nz + 255) / 256, 256, 0, stream>>>(summed1, nz);
  count_kernel<<<(N_EDGES + 255) / 256, 256, 0, stream>>>(dst, cnt);
  wprep_kernel<<<(96 * 64 + 255) / 256, 256, 0, stream>>>(e1w2, w2t1, 96);
  wprep_kernel<<<(1024 * 64 + 255) / 256, 256, 0, stream>>>(e2w2, w2t2, 1024);

  // ---- layer 1 ----
  edge_mlp_kernel<<<(N_EDGES * 64 + 255) / 256, 256, 0, stream>>>(ea, e1w1, e1b1, hid);
  msg_kernel<3><<<512, 128, 0, stream>>>(x, hid, w2t1, e1b2, src, dst, summed1);
  node1_kernel<<<(N_NODES * 32 + 255) / 256, 256, 0, stream>>>(summed1, cnt, x, c1r, c1b, h);

  // ---- layer 2 ----
  edge_mlp_kernel<<<(N_EDGES * 64 + 255) / 256, 256, 0, stream>>>(ea, e2w1, e2b1, hid);
  msg_kernel<32><<<512, 128, 0, stream>>>(h, hid, w2t2, e2b2, src, dst, summed2);

  // ---- node update 2 + fc ----
  node2_kernel<<<(N_NODES + 255) / 256, 256, 0, stream>>>(summed2, cnt, h, c2r, c2b, fcw, fcb, out);
}
